// CRF_63565515981291
// MI455X (gfx1250) — matrix operand order repaired and checked by reading, not compiled, not run on hardware
//
#include <hip/hip_runtime.h>

typedef __attribute__((ext_vector_type(16))) __bf16        v16bf;
typedef __attribute__((ext_vector_type(8)))  float         v8f;
typedef __attribute__((ext_vector_type(8)))  unsigned int  v8u;

#define Bb      256
#define Ss      1024
#define Tt      48
#define TP      64        // T padded to multiple of 32 (WMMA K)
#define START_T 45
#define STOP_T  46

#define LOG2E   1.44269504f
#define LN2     0.69314718f

// LDS strides (elements), padded for conflict-free 64-bank access
#define P_STRIDE  72      // ushort(bf16) rows of P (the linear-domain state)
#define E_STRIDE  74      // ushort(bf16) rows of E^T

// float -> bf16 bits via native convert (v_cvt on gfx1250)
__device__ __forceinline__ unsigned short f2bf(float f) {
  __bf16 h = (__bf16)f;
  return __builtin_bit_cast(unsigned short, h);
}
__device__ __forceinline__ float bf2f(unsigned short h) {
  return __builtin_bit_cast(float, (unsigned int)h << 16);
}

// A fragment: 16x32 bf16, rows in lanes 0..15 (K-halves split across lane-halves).
// lane(l<16): V0..3 hold K=0,1..6,7 ; V4..7 hold K=16..23.  lane(l>=16): +8 / +24.
__device__ __forceinline__ v16bf load_a_frag(const unsigned short* pbuf, int r, int hi, int k0) {
  v8u au;
  const unsigned int* row = (const unsigned int*)(pbuf + r * P_STRIDE);
#pragma unroll
  for (int v = 0; v < 8; ++v) {
    int k = k0 + ((v < 4) ? (8 * hi + 2 * v) : (16 + 8 * hi + 2 * (v - 4)));
    au[v] = row[k >> 1];
  }
  return __builtin_bit_cast(v16bf, au);
}

// B fragment: 32x16 bf16. lane l holds N=n0+l%16; VGPR v packs K like A (8h+2v.. / 16+8h+2(v-4)..).
__device__ __forceinline__ v16bf load_b_frag(const unsigned short* ebuf, int lane, int k0, int n0) {
  v8u bu;
  const int hb = lane >> 4;
  const unsigned short* col = ebuf + n0 + (lane & 15);
#pragma unroll
  for (int v = 0; v < 8; ++v) {
    int k = k0 + ((v < 4) ? (8 * hb + 2 * v) : (16 + 8 * hb + 2 * (v - 4)));
    bu[v] = (unsigned int)col[k * E_STRIDE] | ((unsigned int)col[(k + 1) * E_STRIDE] << 16);
  }
  return __builtin_bit_cast(v16bf, bu);
}

// ---------------- gold path score: one block per batch ----------------
__global__ __launch_bounds__(256) void crf_gold_kernel(
    const float* __restrict__ feats, const float* __restrict__ masks,
    const int* __restrict__ tags, const float* __restrict__ trans,
    float* __restrict__ gold)
{
  const int b   = blockIdx.x;
  const int tid = threadIdx.x;
  __shared__ float redg[8], redl[8];
  float part = 0.f, lpart = 0.f;
  for (int s = tid; s < Ss; s += 256) {
    float mk = masks[b * Ss + s];
    int   tg = tags[b * Ss + s];
    int   pv = (s == 0) ? START_T : tags[b * Ss + s - 1];
    part  += (feats[((long)b * Ss + s) * Tt + tg] + trans[tg * Tt + pv]) * mk;
    lpart += mk;
  }
#pragma unroll
  for (int off = 16; off > 0; off >>= 1) {
    part  += __shfl_xor(part,  off, 32);
    lpart += __shfl_xor(lpart, off, 32);
  }
  int w = tid >> 5;
  if ((tid & 31) == 0) { redg[w] = part; redl[w] = lpart; }
  __syncthreads();
  if (tid == 0) {
    float g = 0.f, L = 0.f;
#pragma unroll
    for (int i = 0; i < 8; ++i) { g += redg[i]; L += redl[i]; }
    int len  = (int)(L + 0.5f);
    int last = (len == 0) ? START_T : tags[b * Ss + len - 1];
    gold[b] = g + trans[STOP_T * Tt + last];
  }
}

// ------------- forward scan: 1 wave per block, 16 batch rows per wave -------------
// State: pbuf[m][i] = exp(score[m,i] - mrow[m]) in bf16 (max-normalized, <=1),
//        per-lane register mreg = running log-scale for the row this lane owns.
// Per step: acc = P x E^T (6 WMMAs); w_i = acc_i * exp(c_i + feat_i);
//           P' = w / max(w); scale += ln(max(w)).
__global__ __launch_bounds__(32) void crf_forward_kernel(
    const float* __restrict__ feats, const float* __restrict__ masks,
    const float* __restrict__ trans, const float* __restrict__ gold,
    float* __restrict__ out)
{
  __shared__ unsigned short pbuf[16][P_STRIDE];      // linear-domain state, bf16
  __shared__ unsigned short ebuf[TP][E_STRIDE];      // E^T[j][i] = exp(trans[i][j]-c_i)
  __shared__ float          cvec[TP];
  __shared__ float          mrow[16];

  const int lane = threadIdx.x;
  const int b0   = blockIdx.x * 16;
  const int n    = lane & 15;    // D column / P row handled by this lane
  const int hi   = lane >> 4;

  // ---- prologue: c_i = rowmax(trans), E^T in bf16, zero-padded to 64 ----
  for (int i = lane; i < Tt; i += 32) {
    float c = -3.0e38f;
    for (int j = 0; j < Tt; ++j) c = fmaxf(c, trans[i * Tt + j]);
    cvec[i] = c;
  }
  for (int i = Tt + lane; i < TP; i += 32) cvec[i] = 0.f;
  for (int idx = lane; idx < 16 * P_STRIDE; idx += 32) (&pbuf[0][0])[idx] = 0;
  for (int idx = lane; idx < TP * E_STRIDE; idx += 32) (&ebuf[0][0])[idx] = 0;
  __syncthreads();
  for (int idx = lane; idx < Tt * Tt; idx += 32) {
    int i = idx / Tt, j = idx - i * Tt;                // i = next, j = prev
    ebuf[j][i] = f2bf(__builtin_amdgcn_exp2f((trans[i * Tt + j] - cvec[i]) * LOG2E));
  }
  __syncthreads();
  if (lane < 16) pbuf[lane][START_T] = 0x3F80;   // bf16(1.0): one-hot at START
  __syncthreads();

  // ---- loop-invariant B fragments: 3 N-tiles x 2 K-chunks ----
  v16bf bfr[3][2];
#pragma unroll
  for (int t = 0; t < 3; ++t)
#pragma unroll
    for (int kc = 0; kc < 2; ++kc)
      bfr[t][kc] = load_b_frag(&ebuf[0][0], lane, kc * 32, t * 16);

  const long   bstride = (long)Ss * Tt;
  const float* fbase   = feats + (long)b0 * bstride;

  float mreg = 0.f;   // running log-scale: lane n<8 owns row (n + 8*hi)

  for (int s = 0; s < Ss; ++s) {
    // (1) g_i = c_i + feat_i for the 24 elements this lane owns (issue loads early)
    float gv[3][8];
    float mk[8];
#pragma unroll
    for (int rr = 0; rr < 8; ++rr) {
      const float* fr = fbase + (long)(rr + 8 * hi) * bstride + (long)s * Tt;
      __builtin_prefetch((const void*)(fr + Tt), 0, 3);   // next step's feats, near-cache
      mk[rr] = masks[(b0 + rr + 8 * hi) * Ss + s];
#pragma unroll
      for (int t = 0; t < 3; ++t) gv[t][rr] = cvec[t * 16 + n] + fr[t * 16 + n];
    }

    // (2) A fragments from current state
    v16bf a0 = load_a_frag(&pbuf[0][0], n, hi, 0);
    v16bf a1 = load_a_frag(&pbuf[0][0], n, hi, 32);

    // (3) 6x v_wmma_f32_16x16x32_bf16; independent K-chunk accumulators
    v8f acc[3];
#pragma unroll
    for (int t = 0; t < 3; ++t) {
      v8f z = {};
      v8f d0 = __builtin_amdgcn_wmma_f32_16x16x32_bf16(false, a0, false, bfr[t][0],
                                                       (short)0, z, false, false);
      v8f d1 = __builtin_amdgcn_wmma_f32_16x16x32_bf16(false, a1, false, bfr[t][1],
                                                       (short)0, z, false, false);
      acc[t] = d0 + d1;
    }

    // (4) w = acc * exp(g); row max across the 16-lane half; renormalize state
    float w[3][8], wmax[8];
#pragma unroll
    for (int rr = 0; rr < 8; ++rr) {
#pragma unroll
      for (int t = 0; t < 3; ++t)
        w[t][rr] = acc[t][rr] * __builtin_amdgcn_exp2f(gv[t][rr] * LOG2E);
      float wm = fmaxf(fmaxf(w[0][rr], w[1][rr]), w[2][rr]);
      wm = fmaxf(wm, __shfl_xor(wm, 1, 32));
      wm = fmaxf(wm, __shfl_xor(wm, 2, 32));
      wm = fmaxf(wm, __shfl_xor(wm, 4, 32));
      wm = fmaxf(wm, __shfl_xor(wm, 8, 32));
      wmax[rr] = fmaxf(wm, 1e-35f);
    }
#pragma unroll
    for (int rr = 0; rr < 8; ++rr) {
      if (mk[rr] > 0.5f) {
        float inv = __builtin_amdgcn_rcpf(wmax[rr]);
#pragma unroll
        for (int t = 0; t < 3; ++t)
          pbuf[rr + 8 * hi][t * 16 + n] = f2bf(w[t][rr] * inv);
        // register-resident scale update: lane n==rr owns row rr+8*hi
        if (n == rr) mreg += LN2 * __builtin_amdgcn_logf(wmax[rr]);
      }
    }
    __syncthreads();
  }

  // spill per-lane scales to LDS for the epilogue
  if (n < 8) mrow[n + 8 * hi] = mreg;
  __syncthreads();

  // ---- epilogue: log_z = mrow + logsumexp(ln(v) + trans[STOP,:]) ; out = log_z - gold ----
  float sl[24];
  float tmax = -3.0e38f;
#pragma unroll
  for (int j = 0; j < 24; ++j) {
    int jj = hi * 24 + j;
    float v = bf2f(pbuf[n][jj]);
    sl[j] = LN2 * __builtin_amdgcn_logf(v) + trans[STOP_T * Tt + jj];
    tmax = fmaxf(tmax, sl[j]);
  }
  tmax = fmaxf(tmax, __shfl_xor(tmax, 16, 32));
  float ssum = 0.f;
#pragma unroll
  for (int j = 0; j < 24; ++j)
    ssum += __builtin_amdgcn_exp2f((sl[j] - tmax) * LOG2E);
  ssum += __shfl_xor(ssum, 16, 32);
  if (hi == 0)
    out[b0 + n] = mrow[n] + tmax + LN2 * __builtin_amdgcn_logf(ssum) - gold[b0 + n];
}

extern "C" void kernel_launch(void* const* d_in, const int* in_sizes, int n_in,
                              void* d_out, int out_size, void* d_ws, size_t ws_size,
                              hipStream_t stream) {
  const float* feats = (const float*)d_in[0];   // [B,S,T]
  const float* masks = (const float*)d_in[1];   // [B,S]
  const int*   tags  = (const int*)d_in[2];     // [B,S]
  const float* trans = (const float*)d_in[3];   // [T,T]
  float* out  = (float*)d_out;                  // [B]
  float* gold = (float*)d_ws;                   // [B] scratch

  crf_gold_kernel<<<Bb, 256, 0, stream>>>(feats, masks, tags, trans, gold);
  crf_forward_kernel<<<Bb / 16, 32, 0, stream>>>(feats, masks, trans, gold, out);
}
